// SchnetNet_83047487635764
// MI455X (gfx1250) — matrix operand order repaired and checked by reading, not compiled, not run on hardware
//
#include <hip/hip_runtime.h>

#define DEV_INLINE __device__ __forceinline__

typedef __attribute__((ext_vector_type(16))) __bf16 v16bf;
typedef __attribute__((ext_vector_type(8)))  __bf16 v8bf;
typedef __attribute__((ext_vector_type(8)))  float  v8f;

union ABFrag { v16bf v; v8bf h[2]; };

// ---- problem constants (match reference) ----
constexpr int N_ATOMS  = 50000;
constexpr int N_EDGES  = 800000;
constexpr int N_GRAPHS = 1000;
constexpr int DIM      = 64;
constexpr int N_RBF    = 300;
constexpr int KPAD     = 320;      // 300 padded to multiple of 32 (zero-filled)
constexpr float RBF_MAX = 30.0f;
constexpr float RBF_DELTA = RBF_MAX / (N_RBF - 1);
constexpr float RBF_COEFF = -0.5f / (RBF_DELTA * RBF_DELTA);
constexpr float LN2 = 0.6931471805599453f;

// ---- helpers ----
DEV_INLINE __bf16 f2bf(float f) {
  union { float f; unsigned u; } x; x.f = f;
  unsigned r = (x.u + 0x7FFFu + ((x.u >> 16) & 1u)) >> 16;   // round-nearest-even
  union { unsigned short s; __bf16 b; } y; y.s = (unsigned short)r;
  return y.b;
}

DEV_INLINE float sspf(float x) {   // shifted softplus, numerically safe
  float ax = fabsf(x);
  return fmaxf(x, 0.0f) + __logf(1.0f + __expf(-ax)) - LN2;
}

// A fragment (16x32 bf16, M x K) from a row-major bf16 tile with row stride `stride`.
// ISA layout: lane L: M = L%16; K halves {0..7,16..23} (L<16) / {8..15,24..31}.
DEV_INLINE v16bf load_a_frag(const __bf16* tile, int stride, int l16, int hi, int k0) {
  const __bf16* row = tile + l16 * stride + k0 + (hi ? 8 : 0);
  ABFrag a;
  a.h[0] = *(const v8bf*)(row);
  a.h[1] = *(const v8bf*)(row + 16);
  return a.v;
}

// B fragment (32x16 bf16, K x N) from an N-major (transposed) bf16 weight with
// row stride `stride` (= padded K). lane L: N = n0 + L%16; K = {0..7,16..23}/{8..15,24..31}.
DEV_INLINE v16bf load_b_frag(const __bf16* wt, int stride, int l16, int hi, int n0, int k0) {
  const __bf16* row = wt + (size_t)(n0 + l16) * stride + k0 + (hi ? 8 : 0);
  ABFrag b;
  b.h[0] = *(const v8bf*)(row);
  b.h[1] = *(const v8bf*)(row + 16);
  return b.v;
}

// ---- small utility kernels ----
__global__ void zero_f32_kernel(float* __restrict__ p, int n) {
  int i = blockIdx.x * blockDim.x + threadIdx.x;
  if (i < n) p[i] = 0.0f;
}

// dst[n*Kpad + k] = bf16( k<K ? src[k*N + n] : 0 )  — N-major transpose+convert
__global__ void twt_kernel(const float* __restrict__ src, __bf16* __restrict__ dst,
                           int K, int N, int Kpad) {
  int i = blockIdx.x * blockDim.x + threadIdx.x;
  int total = N * Kpad;
  if (i < total) {
    int n = i / Kpad;
    int k = i - n * Kpad;
    float v = (k < K) ? src[(size_t)k * N + n] : 0.0f;
    dst[i] = f2bf(v);
  }
}

__global__ void embed_kernel(const int* __restrict__ Z, const float* __restrict__ emb,
                             float* __restrict__ X, int n) {
  int i = blockIdx.x * blockDim.x + threadIdx.x;
  if (i < n) {
    int a = i >> 6, c = i & 63;
    X[i] = emb[(size_t)Z[a] * DIM + c];
  }
}

// ---- atom-wise: h = X @ in2f_W[b]  (one 16-atom tile per wave) ----
__global__ __launch_bounds__(32)
void in2f_kernel(const float* __restrict__ X, const __bf16* __restrict__ Wt,
                 float* __restrict__ h) {
  __shared__ __bf16 sA[16 * DIM];
  const int lane = threadIdx.x, hi = lane >> 4, l16 = lane & 15;
  const int a0 = blockIdx.x * 16;

  for (int t = lane; t < 16 * DIM; t += 32)
    sA[t] = f2bf(X[(size_t)a0 * DIM + t]);
  __syncthreads();

  v8f acc[4];
#pragma unroll
  for (int t = 0; t < 4; ++t)
#pragma unroll
    for (int r = 0; r < 8; ++r) acc[t][r] = 0.0f;

#pragma unroll
  for (int kc = 0; kc < 2; ++kc) {
    v16bf a = load_a_frag(sA, DIM, l16, hi, kc * 32);
#pragma unroll
    for (int t = 0; t < 4; ++t) {
      v16bf b = load_b_frag(Wt, DIM, l16, hi, t * 16, kc * 32);
      acc[t] = __builtin_amdgcn_wmma_f32_16x16x32_bf16(false, a, false, b,
                                                       (short)0, acc[t], false, false);
    }
  }
#pragma unroll
  for (int t = 0; t < 4; ++t)
#pragma unroll
    for (int r = 0; r < 8; ++r)
      h[(size_t)(a0 + r + hi * 8) * DIM + t * 16 + l16] = acc[t][r];
}

// ---- fused edge kernel (32 edges / wave, two M-tiles sharing B fragments):
//      rbf -> WMMA(fW1)+b1 -> ssp -> WMMA(fW2)+b2 -> gather h[idx_j] * W -> scatter-add agg
__global__ __launch_bounds__(32)
void edge_kernel(const int* __restrict__ idx_i, const int* __restrict__ idx_j,
                 const float* __restrict__ dist,
                 const __bf16* __restrict__ W1t, const float* __restrict__ b1,
                 const __bf16* __restrict__ W2t, const float* __restrict__ b2,
                 const float* __restrict__ h, float* __restrict__ agg) {
  __shared__ __bf16 sRbf[32 * KPAD];   // 20 KB
  __shared__ __bf16 sT[32 * DIM];      // 4 KB (post-ssp intermediate)
  __shared__ int   sI[32], sJ[32];
  __shared__ float sD[32];

  const int lane = threadIdx.x, hi = lane >> 4, l16 = lane & 15;
  const int e0 = blockIdx.x * 32;      // N_EDGES % 32 == 0

  sI[lane] = idx_i[e0 + lane];
  sJ[lane] = idx_j[e0 + lane];
  sD[lane] = dist[e0 + lane];
  __syncthreads();

  // compute 32 x KPAD rbf tile (zero padded beyond N_RBF)
  for (int m = 0; m < 32; ++m) {
    float dm = sD[m];
    for (int k = lane; k < KPAD; k += 32) {
      float v = 0.0f;
      if (k < N_RBF) {
        float diff = dm - (float)k * RBF_DELTA;
        v = __expf(RBF_COEFF * diff * diff);
      }
      sRbf[m * KPAD + k] = f2bf(v);
    }
  }
  __syncthreads();

  // GEMM1: [32,320] x [320,64]; each B fragment feeds both M-tiles
  v8f acc[2][4];
#pragma unroll
  for (int mt = 0; mt < 2; ++mt)
#pragma unroll
    for (int t = 0; t < 4; ++t)
#pragma unroll
      for (int r = 0; r < 8; ++r) acc[mt][t][r] = 0.0f;

  for (int kc = 0; kc < KPAD / 32; ++kc) {
    v16bf a0 = load_a_frag(sRbf, KPAD, l16, hi, kc * 32);
    v16bf a1 = load_a_frag(sRbf + 16 * KPAD, KPAD, l16, hi, kc * 32);
#pragma unroll
    for (int t = 0; t < 4; ++t) {
      v16bf b = load_b_frag(W1t, KPAD, l16, hi, t * 16, kc * 32);
      acc[0][t] = __builtin_amdgcn_wmma_f32_16x16x32_bf16(false, a0, false, b,
                                                          (short)0, acc[0][t], false, false);
      acc[1][t] = __builtin_amdgcn_wmma_f32_16x16x32_bf16(false, a1, false, b,
                                                          (short)0, acc[1][t], false, false);
    }
  }

  // bias + ssp -> LDS (C layout: M = mt*16 + r + 8*hi, N = t*16 + l16)
#pragma unroll
  for (int t = 0; t < 4; ++t) {
    float bb = b1[t * 16 + l16];
#pragma unroll
    for (int mt = 0; mt < 2; ++mt)
#pragma unroll
      for (int r = 0; r < 8; ++r)
        sT[(mt * 16 + r + hi * 8) * DIM + t * 16 + l16] = f2bf(sspf(acc[mt][t][r] + bb));
  }
  __syncthreads();

  // GEMM2: [32,64] x [64,64], C initialized with bias b2; shared B fragments
  v8f w[2][4];
#pragma unroll
  for (int t = 0; t < 4; ++t) {
    float bb = b2[t * 16 + l16];
#pragma unroll
    for (int mt = 0; mt < 2; ++mt)
#pragma unroll
      for (int r = 0; r < 8; ++r) w[mt][t][r] = bb;
  }
#pragma unroll
  for (int kc = 0; kc < 2; ++kc) {
    v16bf a0 = load_a_frag(sT, DIM, l16, hi, kc * 32);
    v16bf a1 = load_a_frag(sT + 16 * DIM, DIM, l16, hi, kc * 32);
#pragma unroll
    for (int t = 0; t < 4; ++t) {
      v16bf b = load_b_frag(W2t, DIM, l16, hi, t * 16, kc * 32);
      w[0][t] = __builtin_amdgcn_wmma_f32_16x16x32_bf16(false, a0, false, b,
                                                        (short)0, w[0][t], false, false);
      w[1][t] = __builtin_amdgcn_wmma_f32_16x16x32_bf16(false, a1, false, b,
                                                        (short)0, w[1][t], false, false);
    }
  }

  // msg = h[idx_j] * W ; atomic scatter-add into agg[idx_i]
#pragma unroll
  for (int mt = 0; mt < 2; ++mt)
#pragma unroll
    for (int r = 0; r < 8; ++r) {
      int m = mt * 16 + r + hi * 8;
      const float* hrow = h + (size_t)sJ[m] * DIM;
      float*       arow = agg + (size_t)sI[m] * DIM;
#pragma unroll
      for (int t = 0; t < 4; ++t) {
        int n = t * 16 + l16;
        atomicAdd(&arow[n], w[mt][t][r] * hrow[n]);
      }
    }
}

// ---- atom-wise: X += ssp(agg@o1W + o1b) @ o2W + o2b ----
__global__ __launch_bounds__(32)
void f2out_kernel(const float* __restrict__ agg,
                  const __bf16* __restrict__ o1T, const float* __restrict__ o1b,
                  const __bf16* __restrict__ o2T, const float* __restrict__ o2b,
                  float* __restrict__ X) {
  __shared__ __bf16 sA[16 * DIM];
  __shared__ __bf16 sT[16 * DIM];
  const int lane = threadIdx.x, hi = lane >> 4, l16 = lane & 15;
  const int a0 = blockIdx.x * 16;      // N_ATOMS % 16 == 0

  for (int t = lane; t < 16 * DIM; t += 32)
    sA[t] = f2bf(agg[(size_t)a0 * DIM + t]);
  __syncthreads();

  v8f acc[4];
#pragma unroll
  for (int t = 0; t < 4; ++t) {
    float bb = o1b[t * 16 + l16];
#pragma unroll
    for (int r = 0; r < 8; ++r) acc[t][r] = bb;
  }
#pragma unroll
  for (int kc = 0; kc < 2; ++kc) {
    v16bf a = load_a_frag(sA, DIM, l16, hi, kc * 32);
#pragma unroll
    for (int t = 0; t < 4; ++t) {
      v16bf b = load_b_frag(o1T, DIM, l16, hi, t * 16, kc * 32);
      acc[t] = __builtin_amdgcn_wmma_f32_16x16x32_bf16(false, a, false, b,
                                                       (short)0, acc[t], false, false);
    }
  }
#pragma unroll
  for (int t = 0; t < 4; ++t)
#pragma unroll
    for (int r = 0; r < 8; ++r)
      sT[(r + hi * 8) * DIM + t * 16 + l16] = f2bf(sspf(acc[t][r]));
  __syncthreads();

  v8f v[4];
#pragma unroll
  for (int t = 0; t < 4; ++t) {
    float bb = o2b[t * 16 + l16];
#pragma unroll
    for (int r = 0; r < 8; ++r) v[t][r] = bb;
  }
#pragma unroll
  for (int kc = 0; kc < 2; ++kc) {
    v16bf a = load_a_frag(sT, DIM, l16, hi, kc * 32);
#pragma unroll
    for (int t = 0; t < 4; ++t) {
      v16bf b = load_b_frag(o2T, DIM, l16, hi, t * 16, kc * 32);
      v[t] = __builtin_amdgcn_wmma_f32_16x16x32_bf16(false, a, false, b,
                                                     (short)0, v[t], false, false);
    }
  }
#pragma unroll
  for (int t = 0; t < 4; ++t)
#pragma unroll
    for (int r = 0; r < 8; ++r) {
      size_t o = (size_t)(a0 + r + hi * 8) * DIM + t * 16 + l16;
      X[o] = X[o] + v[t][r];
    }
}

// ---- output head: ssp(X@outW1)@outW2 + b -> segment-sum into energies ----
__global__ __launch_bounds__(32)
void out_kernel(const float* __restrict__ X, const __bf16* __restrict__ W1t,
                const float* __restrict__ w2, const float* __restrict__ b2,
                const int* __restrict__ graph_idx, float* __restrict__ energies) {
  __shared__ __bf16 sA[16 * DIM];
  const int lane = threadIdx.x, hi = lane >> 4, l16 = lane & 15;
  const int a0 = blockIdx.x * 16;

  for (int t = lane; t < 16 * DIM; t += 32)
    sA[t] = f2bf(X[(size_t)a0 * DIM + t]);
  __syncthreads();

  v8f acc[2];
#pragma unroll
  for (int t = 0; t < 2; ++t)
#pragma unroll
    for (int r = 0; r < 8; ++r) acc[t][r] = 0.0f;

#pragma unroll
  for (int kc = 0; kc < 2; ++kc) {
    v16bf a = load_a_frag(sA, DIM, l16, hi, kc * 32);
#pragma unroll
    for (int t = 0; t < 2; ++t) {
      v16bf b = load_b_frag(W1t, DIM, l16, hi, t * 16, kc * 32);
      acc[t] = __builtin_amdgcn_wmma_f32_16x16x32_bf16(false, a, false, b,
                                                       (short)0, acc[t], false, false);
    }
  }

  float w2a = w2[l16], w2b = w2[16 + l16], bias = b2[0];
#pragma unroll
  for (int r = 0; r < 8; ++r) {
    float p = sspf(acc[0][r]) * w2a + sspf(acc[1][r]) * w2b;
    // reduce over the 16 lanes sharing the same M row (stays inside each half)
    p += __shfl_xor(p, 1, 32);
    p += __shfl_xor(p, 2, 32);
    p += __shfl_xor(p, 4, 32);
    p += __shfl_xor(p, 8, 32);
    if (l16 == 0) {
      int m = r + hi * 8;
      atomicAdd(&energies[graph_idx[a0 + m]], p + bias);
    }
  }
}

// ---------------- host launcher ----------------
extern "C" void kernel_launch(void* const* d_in, const int* in_sizes, int n_in,
                              void* d_out, int out_size, void* d_ws, size_t ws_size,
                              hipStream_t stream) {
  const int*   Z        = (const int*)  d_in[0];
  const int*   idx_i    = (const int*)  d_in[1];
  const int*   idx_j    = (const int*)  d_in[2];
  const float* dist     = (const float*)d_in[3];
  const int*   graphIdx = (const int*)  d_in[4];
  const float* emb      = (const float*)d_in[5];
  const float* in2f_W   = (const float*)d_in[6];
  const float* fW1      = (const float*)d_in[7];
  const float* fb1      = (const float*)d_in[8];
  const float* fW2      = (const float*)d_in[9];
  const float* fb2      = (const float*)d_in[10];
  const float* o1W      = (const float*)d_in[11];
  const float* o1b      = (const float*)d_in[12];
  const float* o2W      = (const float*)d_in[13];
  const float* o2b      = (const float*)d_in[14];
  const float* outW1    = (const float*)d_in[15];
  const float* outW2    = (const float*)d_in[16];
  const float* outb2    = (const float*)d_in[17];
  float* energies = (float*)d_out;

  // workspace carve-up
  char* ws = (char*)d_ws;
  size_t off = 0;
  float* X   = (float*)(ws + off); off += (size_t)N_ATOMS * DIM * sizeof(float);
  float* h   = (float*)(ws + off); off += (size_t)N_ATOMS * DIM * sizeof(float);
  float* agg = (float*)(ws + off); off += (size_t)N_ATOMS * DIM * sizeof(float);
  __bf16* W1t    = (__bf16*)(ws + off); off += (size_t)3 * DIM * KPAD * 2;
  __bf16* W2t    = (__bf16*)(ws + off); off += (size_t)3 * DIM * DIM * 2;
  __bf16* in2fT  = (__bf16*)(ws + off); off += (size_t)3 * DIM * DIM * 2;
  __bf16* o1T    = (__bf16*)(ws + off); off += (size_t)3 * DIM * DIM * 2;
  __bf16* o2T    = (__bf16*)(ws + off); off += (size_t)3 * DIM * DIM * 2;
  __bf16* oW1t   = (__bf16*)(ws + off); off += (size_t)32 * DIM * 2;

  // 1) weight transpose + bf16 convert (tiny, L2-resident thereafter)
  for (int b = 0; b < 3; ++b) {
    int c1 = DIM * KPAD, c2 = DIM * DIM;
    twt_kernel<<<(c1 + 255) / 256, 256, 0, stream>>>(fW1 + (size_t)b * N_RBF * DIM,
                                                     W1t + (size_t)b * DIM * KPAD,
                                                     N_RBF, DIM, KPAD);
    twt_kernel<<<(c2 + 255) / 256, 256, 0, stream>>>(fW2 + (size_t)b * DIM * DIM,
                                                     W2t + (size_t)b * DIM * DIM,
                                                     DIM, DIM, DIM);
    twt_kernel<<<(c2 + 255) / 256, 256, 0, stream>>>(in2f_W + (size_t)b * DIM * DIM,
                                                     in2fT + (size_t)b * DIM * DIM,
                                                     DIM, DIM, DIM);
    twt_kernel<<<(c2 + 255) / 256, 256, 0, stream>>>(o1W + (size_t)b * DIM * DIM,
                                                     o1T + (size_t)b * DIM * DIM,
                                                     DIM, DIM, DIM);
    twt_kernel<<<(c2 + 255) / 256, 256, 0, stream>>>(o2W + (size_t)b * DIM * DIM,
                                                     o2T + (size_t)b * DIM * DIM,
                                                     DIM, DIM, DIM);
  }
  twt_kernel<<<(32 * DIM + 255) / 256, 256, 0, stream>>>(outW1, oW1t, DIM, 32, DIM);

  // 2) embedding gather
  embed_kernel<<<(N_ATOMS * DIM + 255) / 256, 256, 0, stream>>>(Z, emb, X, N_ATOMS * DIM);

  // 3) interaction blocks
  const int atomTiles = N_ATOMS / 16;   // 3125
  const int edgeTiles = N_EDGES / 32;   // 25000 (32 edges per wave)
  for (int b = 0; b < 3; ++b) {
    in2f_kernel<<<atomTiles, 32, 0, stream>>>(X, in2fT + (size_t)b * DIM * DIM, h);
    zero_f32_kernel<<<(N_ATOMS * DIM + 255) / 256, 256, 0, stream>>>(agg, N_ATOMS * DIM);
    edge_kernel<<<edgeTiles, 32, 0, stream>>>(idx_i, idx_j, dist,
                                              W1t + (size_t)b * DIM * KPAD, fb1 + (size_t)b * DIM,
                                              W2t + (size_t)b * DIM * DIM, fb2 + (size_t)b * DIM,
                                              h, agg);
    f2out_kernel<<<atomTiles, 32, 0, stream>>>(agg,
                                               o1T + (size_t)b * DIM * DIM, o1b + (size_t)b * DIM,
                                               o2T + (size_t)b * DIM * DIM, o2b + (size_t)b * DIM,
                                               X);
  }

  // 4) output head + per-graph segment sum
  zero_f32_kernel<<<(N_GRAPHS + 255) / 256, 256, 0, stream>>>(energies, N_GRAPHS);
  out_kernel<<<atomTiles, 32, 0, stream>>>(X, oW1t, outW2, outb2, graphIdx, energies);
}
